// LinearSelfAttention_50010599194737
// MI455X (gfx1250) — matrix operand order repaired and checked by reading, not compiled, not run on hardware
//
#include <hip/hip_runtime.h>

// ---------------------------------------------------------------------------
// LinearSelfAttention for MI455X (gfx1250, wave32, WMMA).
// All matmuls use v_wmma_f32_16x16x32_f16 (f32 data cvt to f16, f32 accum).
// Memory-bound workload: qkv intermediate kept in f16 (201 MB), q-softmax is
// fused into the qkv GEMM epilogue (M-blocks align with head boundaries),
// v is consumed directly from the f16 qkv buffer, k-softmax runs in place.
// Workspace: ~270 MB, with later buffers aliasing the dead qkv region.
// ---------------------------------------------------------------------------

typedef __attribute__((ext_vector_type(16))) _Float16 v16h;
typedef __attribute__((ext_vector_type(8)))  _Float16 v8h;
typedef __attribute__((ext_vector_type(8)))  float    v8f;

union FragH { v16h v; v8h h[2]; };

#define WMMA_F16(A, B, C) \
  __builtin_amdgcn_wmma_f32_16x16x32_f16(false, (A), false, (B), (short)0, (C), false, false)

__device__ __forceinline__ v8h cvt8(float4 a, float4 b) {
  v8h h;
  h[0] = (_Float16)a.x; h[1] = (_Float16)a.y; h[2] = (_Float16)a.z; h[3] = (_Float16)a.w;
  h[4] = (_Float16)b.x; h[5] = (_Float16)b.y; h[6] = (_Float16)b.z; h[7] = (_Float16)b.w;
  return h;
}

// A-fragment (16x32 f16) loaded directly from a row-major f32 matrix.
// Lane L (r=L&15, hi=L>>4): row, K {k0+16*hi .. +15} contiguous.
template<int LDA>
__device__ __forceinline__ v16h load_afrag(const float* __restrict__ A, int row,
                                           int k0, int hi) {
  const float* ap = A + (size_t)row * LDA + k0 + (hi << 4);
  FragH fr;
  fr.h[0] = cvt8(*(const float4*)ap,        *(const float4*)(ap + 4));
  fr.h[1] = cvt8(*(const float4*)(ap + 8),  *(const float4*)(ap + 12));
  return fr.v;
}

// ---------------------------------------------------------------------------
// qkv GEMM + fused q-softmax.
//   C16[b](1536x4096 f16) = w_qkv(1536x256) * x[b](256x4096)
// Block 256 thr = 8 waves (4 M x 2 N); block tile 128x64; K-step 32.
// A: direct per-lane global loads (hot in L2). B: f32->f16 LDS transpose in
// fragment order. M-blocks 0..3 are the q channels (exactly 2 heads / block):
// their epilogue does the d-softmax via a 32 KB LDS tile and writes the
// transposed q16t[(b*8+h)*4096+p][d]; the q third never touches global f32.
// M-blocks 4..11 (k,v channels) store f16 directly into qkv16.
// ---------------------------------------------------------------------------
__global__ __launch_bounds__(256)
void qkv_gemm(const float* __restrict__ A, const float* __restrict__ Bm,
              _Float16* __restrict__ C16, _Float16* __restrict__ q16t)
{
  constexpr int K = 256, N = 4096;
  __shared__ _Float16 lds_b[32 * 64];    // 4 KB, fragment-ordered
  __shared__ float    lds_q[128 * 64];   // 32 KB, q-softmax exchange

  const int tid = threadIdx.x, lane = tid & 31, wave = tid >> 5;
  const int wm = wave & 3, wn = wave >> 2;
  const int hi = lane >> 4, ln = lane & 15;
  const int m0 = blockIdx.y * 128, n0 = blockIdx.x * 64;

  const float* Bb = Bm + (size_t)blockIdx.z * K * N;

  // Per-thread B-fill coordinates (invariant over k-steps):
  const int fk   = tid >> 4;             // k-row 0..15 (chunk1: +16)
  const int fcol = (tid & 15) << 2;      // 4 consecutive columns

  v8f acc[2][2];
  #pragma unroll
  for (int ti = 0; ti < 2; ++ti)
    #pragma unroll
    for (int tj = 0; tj < 2; ++tj)
      #pragma unroll
      for (int i = 0; i < 8; ++i) acc[ti][tj][i] = 0.0f;

  for (int k0 = 0; k0 < K; k0 += 32) {
    __syncthreads();
    {  // stage B tile (32 k x 64 n) into fragment order, straight-line
      const float4 f0 = *(const float4*)(Bb + (size_t)(k0 + fk) * N + n0 + fcol);
      const float4 f1 = *(const float4*)(Bb + (size_t)(k0 + fk + 16) * N + n0 + fcol);
      const float a0[4] = { f0.x, f0.y, f0.z, f0.w };
      const float a1[4] = { f1.x, f1.y, f1.z, f1.w };
      #pragma unroll
      for (int j = 0; j < 4; ++j) {
        const int cc = fcol + j;
        const int base = ((cc >> 4) * 32 + (cc & 15)) << 4;
        lds_b[base + fk]             = (_Float16)a0[j];   // k 0..15  -> lanes 0-15
        lds_b[base + (16 << 4) + fk] = (_Float16)a1[j];   // k 16..31 -> lanes 16-31
      }
    }
    // prefetch next B tile (32 rows x 256B): one 32B granule per thread
    if (k0 + 32 < K)
      __builtin_prefetch(Bb + (size_t)(k0 + 32 + (tid >> 3)) * N + n0 + ((tid & 7) << 3), 0, 1);
    __syncthreads();

    const v16h a0 = load_afrag<K>(A, m0 + wm * 32 + ln,      k0, hi);
    const v16h a1 = load_afrag<K>(A, m0 + wm * 32 + 16 + ln, k0, hi);
    const v16h b0 = *(const v16h*)&lds_b[(((wn * 2 + 0) * 32 + lane) << 4)];
    const v16h b1 = *(const v16h*)&lds_b[(((wn * 2 + 1) * 32 + lane) << 4)];

    acc[0][0] = WMMA_F16(a0, b0, acc[0][0]);
    acc[0][1] = WMMA_F16(a0, b1, acc[0][1]);
    acc[1][0] = WMMA_F16(a1, b0, acc[1][0]);
    acc[1][1] = WMMA_F16(a1, b1, acc[1][1]);
  }

  if (m0 < 512) {
    // ---- q path: fused softmax over d (this block holds 2 complete heads) ----
    #pragma unroll
    for (int ti = 0; ti < 2; ++ti)
      #pragma unroll
      for (int tj = 0; tj < 2; ++tj) {
        const int coll = wn * 32 + tj * 16 + ln;
        #pragma unroll
        for (int r = 0; r < 8; ++r) {
          const int rowl = wm * 32 + ti * 16 + r + (hi << 3);
          lds_q[rowl * 64 + coll] = acc[ti][tj][r];
        }
      }
    __syncthreads();
    if (tid < 128) {                       // one (head, column) per thread
      const int head = tid >> 6, col = tid & 63;
      float v[64];
      float mx = -1e30f;
      #pragma unroll
      for (int d = 0; d < 64; ++d) {
        v[d] = lds_q[(head * 64 + d) * 64 + col];
        mx = fmaxf(mx, v[d]);
      }
      float s = 0.0f;
      #pragma unroll
      for (int d = 0; d < 64; ++d) { v[d] = __expf(v[d] - mx); s += v[d]; }
      const float inv = 0.125f / s;        // includes 1/sqrt(DIM_HEAD)
      const int hg = blockIdx.y * 2 + head;
      _Float16* dst = q16t + ((size_t)(blockIdx.z * 8 + hg) * 4096 + (n0 + col)) * 64;
      #pragma unroll
      for (int c8 = 0; c8 < 8; ++c8) {
        v8h hv;
        #pragma unroll
        for (int j = 0; j < 8; ++j) hv[j] = (_Float16)(v[c8 * 8 + j] * inv);
        *(v8h*)(dst + c8 * 8) = hv;
      }
    }
  } else {
    // ---- k / v path: store f16 directly ----
    _Float16* Cb = C16 + (size_t)blockIdx.z * 1536 * N;
    #pragma unroll
    for (int ti = 0; ti < 2; ++ti)
      #pragma unroll
      for (int tj = 0; tj < 2; ++tj) {
        const int colg = n0 + wn * 32 + tj * 16 + ln;
        #pragma unroll
        for (int r = 0; r < 8; ++r) {
          const int rowg = m0 + wm * 32 + ti * 16 + r + (hi << 3);
          Cb[(size_t)rowg * N + colg] = (_Float16)acc[ti][tj][r];
        }
      }
  }
}

// ---------------------------------------------------------------------------
// k softmax over n, IN PLACE on f16 rows (channels 512..1023 of qkv16).
// One 256-thread block per row; 16 contiguous f16 per thread (v8h x2).
// ---------------------------------------------------------------------------
__global__ __launch_bounds__(256)
void k_softmax(_Float16* __restrict__ qkv16)
{
  __shared__ float red[256];
  const int c = blockIdx.x, b = blockIdx.y, tid = threadIdx.x;
  _Float16* row = qkv16 + ((size_t)b * 1536 + 512 + c) * 4096 + tid * 16;
  v8h h0 = *(const v8h*)row;
  v8h h1 = *(const v8h*)(row + 8);
  float v[16];
  float m = -1e30f;
  #pragma unroll
  for (int i = 0; i < 8; ++i) {
    v[i]     = (float)h0[i];
    v[8 + i] = (float)h1[i];
  }
  #pragma unroll
  for (int i = 0; i < 16; ++i) m = fmaxf(m, v[i]);
  red[tid] = m; __syncthreads();
  for (int st = 128; st > 0; st >>= 1) {
    if (tid < st) red[tid] = fmaxf(red[tid], red[tid + st]);
    __syncthreads();
  }
  m = red[0]; __syncthreads();
  float s = 0.0f;
  #pragma unroll
  for (int i = 0; i < 16; ++i) { v[i] = __expf(v[i] - m); s += v[i]; }
  red[tid] = s; __syncthreads();
  for (int st = 128; st > 0; st >>= 1) {
    if (tid < st) red[tid] += red[tid + st];
    __syncthreads();
  }
  const float inv = 1.0f / red[0];
  #pragma unroll
  for (int i = 0; i < 8; ++i) {
    h0[i] = (_Float16)(v[i] * inv);
    h1[i] = (_Float16)(v[8 + i] * inv);
  }
  *(v8h*)row       = h0;
  *(v8h*)(row + 8) = h1;
}

// ---------------------------------------------------------------------------
// context[d][e] = sum_n k[d][n] * v[e][n], 64x64 per (b,h); K-dim = n = 4096.
// k and v read straight out of qkv16 (channels 512+.. / 1024+..); fragments
// loaded directly from global (both n-contiguous). Output TRANSPOSED f16.
// Block = 128 thr (4 waves); wave w owns rows d in [w*16, w*16+16).
// ---------------------------------------------------------------------------
__global__ __launch_bounds__(128)
void context_gemm(const _Float16* __restrict__ qkv16, _Float16* __restrict__ ctxT)
{
  const int bh = blockIdx.x;              // b*8 + h
  const int b = bh >> 3, h = bh & 7;
  const int tid  = threadIdx.x;
  const int lane = tid & 31, wave = tid >> 5;
  const int hi = lane >> 4, ln = lane & 15;
  const _Float16* kb = qkv16 + ((size_t)b * 1536 + 512  + h * 64) * 4096;
  const _Float16* vb = qkv16 + ((size_t)b * 1536 + 1024 + h * 64) * 4096;
  const int d0 = wave * 16;

  v8f acc[4];
  #pragma unroll
  for (int e = 0; e < 4; ++e)
    #pragma unroll
    for (int i = 0; i < 8; ++i) acc[e][i] = 0.0f;

  for (int n0 = 0; n0 < 4096; n0 += 32) {
    FragH a;  // A-frag: row d0+ln, 16 consecutive K at n0 + 16*hi
    const _Float16* ap = kb + (size_t)(d0 + ln) * 4096 + n0 + (hi << 4);
    a.h[0] = *(const v8h*)ap;
    a.h[1] = *(const v8h*)(ap + 8);
    #pragma unroll
    for (int ej = 0; ej < 4; ++ej) {
      // B-frag: col e = ej*16+ln, 16 consecutive K at n0 + 16*hi (32B load)
      const v16h bv = *(const v16h*)(vb + (size_t)(ej * 16 + ln) * 4096 + n0 + (hi << 4));
      acc[ej] = WMMA_F16(a.v, bv, acc[ej]);
    }
  }
  _Float16* cb = ctxT + (size_t)bh * 4096;
  #pragma unroll
  for (int ej = 0; ej < 4; ++ej)
    #pragma unroll
    for (int r = 0; r < 8; ++r) {
      const int d = d0 + r + (hi << 3);
      const int e = ej * 16 + ln;
      cb[e * 64 + d] = (_Float16)acc[ej][r];   // transposed store
    }
}

// ---------------------------------------------------------------------------
// attn[e][p] = sum_d ctxT[e][d] * q16t[p][d];  M=64, K=64, N=4096 per (b,h).
// Direct global fragment loads; output TRANSPOSED f16 into attnT[b][p][c],
// c = h*64+e: D-frag rows are 8 consecutive e -> one 16B v8h store per tile.
// Block = 256 thr (8 waves = 4 M-tiles x 2 N-halves), block tile 64x128.
// ---------------------------------------------------------------------------
__global__ __launch_bounds__(256)
void attn_gemm(const _Float16* __restrict__ ctxT, const _Float16* __restrict__ q16t,
               _Float16* __restrict__ attnT)
{
  const int bh = blockIdx.y;
  const int p0 = blockIdx.x * 128;
  const int tid = threadIdx.x, lane = tid & 31, wave = tid >> 5;
  const int wm = wave >> 1, wn = wave & 1;
  const int hi = lane >> 4, ln = lane & 15;
  const _Float16* cb = ctxT + (size_t)bh * 4096;
  const _Float16* qb = q16t + (size_t)bh * 4096 * 64;

  v8f acc[4];
  #pragma unroll
  for (int t = 0; t < 4; ++t)
    #pragma unroll
    for (int i = 0; i < 8; ++i) acc[t][i] = 0.0f;

  #pragma unroll
  for (int d0 = 0; d0 < 64; d0 += 32) {
    FragH a;  // A-frag: row e = wm*16+ln
    const _Float16* ap = cb + (wm * 16 + ln) * 64 + d0 + (hi << 4);
    a.h[0] = *(const v8h*)ap;
    a.h[1] = *(const v8h*)(ap + 8);
    #pragma unroll
    for (int tj = 0; tj < 4; ++tj) {
      const int p = p0 + wn * 64 + tj * 16 + ln;
      const v16h bv = *(const v16h*)(qb + (size_t)p * 64 + d0 + (hi << 4));
      acc[tj] = WMMA_F16(a.v, bv, acc[tj]);
    }
  }
  // attnT[b][p][512], channel c = h*64 + e
  _Float16* ob = attnT + (size_t)(bh >> 3) * 4096 * 512 + (size_t)(bh & 7) * 64;
  #pragma unroll
  for (int tj = 0; tj < 4; ++tj) {
    const int p = p0 + wn * 64 + tj * 16 + ln;
    v8h hv;
    #pragma unroll
    for (int r = 0; r < 8; ++r) hv[r] = (_Float16)acc[tj][r];
    *(v8h*)(ob + (size_t)p * 512 + wm * 16 + (hi << 3)) = hv;
  }
}

// ---------------------------------------------------------------------------
// out GEMM: pre[b](256x4096) = w_out(256x512) * attnT[b]^T + b_out, no LDS:
// A-frags direct from f32 global (cvt), B-frags direct 32B loads from attnT.
// Also emits deterministic per-block (sum, sumsq) partials for GroupNorm.
// ---------------------------------------------------------------------------
__global__ __launch_bounds__(256)
void out_gemm(const float* __restrict__ A, const _Float16* __restrict__ Bt,
              const float* __restrict__ bias, float* __restrict__ Cm,
              float* __restrict__ partials)
{
  constexpr int K = 512, N = 4096, M = 256;
  __shared__ float red[512];
  const int tid = threadIdx.x, lane = tid & 31, wave = tid >> 5;
  const int wm = wave & 3, wn = wave >> 2;
  const int hi = lane >> 4, ln = lane & 15;
  const int m0 = blockIdx.y * 128, n0 = blockIdx.x * 64;

  const _Float16* Bb = Bt + (size_t)blockIdx.z * N * K;
  float*          Cb = Cm + (size_t)blockIdx.z * M * N;

  const _Float16* bp0 = Bb + (size_t)(n0 + wn * 32 + ln) * K + (hi << 4);
  const _Float16* bp1 = Bb + (size_t)(n0 + wn * 32 + 16 + ln) * K + (hi << 4);

  v8f acc[2][2];
  #pragma unroll
  for (int ti = 0; ti < 2; ++ti)
    #pragma unroll
    for (int tj = 0; tj < 2; ++tj)
      #pragma unroll
      for (int i = 0; i < 8; ++i) acc[ti][tj][i] = 0.0f;

  #pragma unroll 2
  for (int k0 = 0; k0 < K; k0 += 32) {
    const v16h a0 = load_afrag<K>(A, m0 + wm * 32 + ln,      k0, hi);
    const v16h a1 = load_afrag<K>(A, m0 + wm * 32 + 16 + ln, k0, hi);
    const v16h b0 = *(const v16h*)(bp0 + k0);
    const v16h b1 = *(const v16h*)(bp1 + k0);
    acc[0][0] = WMMA_F16(a0, b0, acc[0][0]);
    acc[0][1] = WMMA_F16(a0, b1, acc[0][1]);
    acc[1][0] = WMMA_F16(a1, b0, acc[1][0]);
    acc[1][1] = WMMA_F16(a1, b1, acc[1][1]);
  }

  float s = 0.0f, s2 = 0.0f;
  #pragma unroll
  for (int ti = 0; ti < 2; ++ti)
    #pragma unroll
    for (int tj = 0; tj < 2; ++tj) {
      const int colg = n0 + wn * 32 + tj * 16 + ln;
      #pragma unroll
      for (int r = 0; r < 8; ++r) {
        const int rowg = m0 + wm * 32 + ti * 16 + r + (hi << 3);
        const float v = acc[ti][tj][r] + bias[rowg];
        Cb[(size_t)rowg * N + colg] = v;
        s += v; s2 += v * v;
      }
    }

  red[tid] = s; red[256 + tid] = s2;
  __syncthreads();
  for (int st = 128; st > 0; st >>= 1) {
    if (tid < st) { red[tid] += red[tid + st]; red[256 + tid] += red[256 + tid + st]; }
    __syncthreads();
  }
  if (tid == 0) {
    const int pidx = (blockIdx.z * gridDim.y + blockIdx.y) * gridDim.x + blockIdx.x;
    partials[pidx * 2]     = red[0];
    partials[pidx * 2 + 1] = red[256];
  }
}

// ---------------------------------------------------------------------------
// Reduce per-block partial (sum, sumsq) -> per-batch (mean, 1/sqrt(var+eps)).
// ---------------------------------------------------------------------------
__global__ __launch_bounds__(128)
void reduce_stats(const float* __restrict__ partials, float* __restrict__ stats)
{
  __shared__ float s1[128], s2[128];
  const int b = blockIdx.x, t = threadIdx.x;
  s1[t] = partials[(b * 128 + t) * 2];
  s2[t] = partials[(b * 128 + t) * 2 + 1];
  __syncthreads();
  for (int st = 64; st > 0; st >>= 1) {
    if (t < st) { s1[t] += s1[t + st]; s2[t] += s2[t + st]; }
    __syncthreads();
  }
  if (t == 0) {
    const float cnt = 256.0f * 4096.0f;
    const float mean = s1[0] / cnt;
    const float var  = s2[0] / cnt - mean * mean;
    stats[b * 2]     = mean;
    stats[b * 2 + 1] = rsqrtf(var + 1e-5f);
  }
}

// ---------------------------------------------------------------------------
// GroupNorm apply: out = (pre - mean)*invstd*gn_w[o] + gn_b[o]
// ---------------------------------------------------------------------------
__global__ __launch_bounds__(256)
void groupnorm_out(const float* __restrict__ pre, const float* __restrict__ stats,
                   const float* __restrict__ gnw, const float* __restrict__ gnb,
                   float* __restrict__ out)
{
  const size_t idx = ((size_t)blockIdx.x * 256 + threadIdx.x) * 4;
  const int b = (int)(idx >> 20);          // / (256*4096)
  const int o = (int)((idx >> 12) & 255);  // channel
  const float mean = stats[b * 2], inv = stats[b * 2 + 1];
  const float w = gnw[o], bb = gnb[o];
  float4 f = *(const float4*)(pre + idx);
  f.x = (f.x - mean) * inv * w + bb;
  f.y = (f.y - mean) * inv * w + bb;
  f.z = (f.z - mean) * inv * w + bb;
  f.w = (f.w - mean) * inv * w + bb;
  *(float4*)(out + idx) = f;
}

// ---------------------------------------------------------------------------
extern "C" void kernel_launch(void* const* d_in, const int* in_sizes, int n_in,
                              void* d_out, int out_size, void* d_ws, size_t ws_size,
                              hipStream_t stream)
{
  const float* x     = (const float*)d_in[0];  // [16,256,64,64]
  const float* w_qkv = (const float*)d_in[1];  // [1536,256]
  const float* w_out = (const float*)d_in[2];  // [256,512]
  const float* b_out = (const float*)d_in[3];  // [256]
  const float* gn_w  = (const float*)d_in[4];  // [256]
  const float* gn_b  = (const float*)d_in[5];  // [256]
  float* out = (float*)d_out;

  // Workspace carve (~270 MB). qkv16 (k,v channels) is dead after
  // context_gemm, so attnT / pre alias it (stream-ordered, safe).
  char* ws = (char*)d_ws;
  _Float16* qkv16 = (_Float16*)(ws + 0);           // 201,326,592 B
  _Float16* attnT = (_Float16*)(ws + 0);           //  67,108,864 B (aliases qkv16)
  float*    pre   = (float*)(ws + 67108864);       //  67,108,864 B (aliases qkv16)
  _Float16* q16t  = (_Float16*)(ws + 201326592);   //  67,108,864 B
  _Float16* ctxT  = (_Float16*)(ws + 268435456);   //   1,048,576 B
  float*    part  = (float*)(ws + 269484032);      //      16,384 B
  float*    stats = (float*)(ws + 269500416);      //         128 B

  // 1) qkv16 = w_qkv @ x[b] (f16 out); q third: fused softmax -> q16t
  qkv_gemm<<<dim3(64, 12, 16), 256, 0, stream>>>(w_qkv, x, qkv16, q16t);
  // 2) softmax k over n, in place on f16 rows
  k_softmax<<<dim3(512, 16), 256, 0, stream>>>(qkv16);
  // 3) context = k @ v^T (64x64 per (b,h)), stored transposed f16
  context_gemm<<<dim3(128), 128, 0, stream>>>(qkv16, ctxT);
  // 4) attn = ctx^T @ q (64x4096 per (b,h)), stored transposed f16 [b][p][c]
  attn_gemm<<<dim3(32, 128), 256, 0, stream>>>(ctxT, q16t, attnT);
  // 5) pre = w_out @ attn[b] + b_out, with per-block stat partials
  out_gemm<<<dim3(64, 2, 16), 256, 0, stream>>>(w_out, attnT, b_out, pre, part);
  // 6) per-batch mean / invstd
  reduce_stats<<<dim3(16), 128, 0, stream>>>(part, stats);
  // 7) GroupNorm apply -> d_out
  groupnorm_out<<<dim3(16384), 256, 0, stream>>>(pre, stats, gn_w, gn_b, out);
}
